// MPNN_40802189312200
// MI455X (gfx1250) — matrix operand order repaired and checked by reading, not compiled, not run on hardware
//
#include <hip/hip_runtime.h>
#include <hip/hip_bf16.h>

#define N_NODES 50000
#define N_EDGES 800000
#define N_GRAPHS 64
#define C 64
#define NC (N_NODES * C)
#define E_TOT (N_EDGES + N_NODES)   // 850000
#define NTILES (E_TOT / 16)         // 53125 (exact)
#define EDGE_BLOCKS 1024            // fixed grid, grid-stride over tiles

typedef __attribute__((ext_vector_type(16))) __bf16 v16bf;
typedef __attribute__((ext_vector_type(8)))  __bf16 v8bf;
typedef __attribute__((ext_vector_type(8)))  float  v8f;

// ---------------- small utility kernels ----------------

__global__ void __launch_bounds__(256) k_zero(float* __restrict__ p, int n) {
    int i = blockIdx.x * 256 + threadIdx.x;
    if (i < n) p[i] = 0.0f;
}

__global__ void __launch_bounds__(256) k_cvt(const float* __restrict__ x,
                                             __bf16* __restrict__ y, int n) {
    int i = blockIdx.x * 256 + threadIdx.x;
    if (i < n) y[i] = (__bf16)x[i];
}

__global__ void __launch_bounds__(256) k_relu_cvt(const float* __restrict__ x,
                                                  __bf16* __restrict__ y, int n) {
    int i = blockIdx.x * 256 + threadIdx.x;
    if (i < n) {
        float v = x[i];
        y[i] = (__bf16)(v > 0.0f ? v : 0.0f);
    }
}

// ---------------- core edge-tile MPNN layer ----------------
// Each wave grid-strides over 16-edge tiles, software-pipelined: the next
// tile's edge-index loads and feature gathers are issued while the current
// tile's GEMM2 + atomic scatter execute. W1 B-fragments are loop-invariant
// registers (GEMM1 WMMAs run back-to-back); W2 B-fragments stream from LDS,
// double-buffered across the GEMM2 nt loop so ds loads overlap WMMAs.
// Divergent addressing uses unsigned 32-bit offsets (SADDR+voffset form).

__global__ void __launch_bounds__(256) k_edges(
    const __bf16* __restrict__ xin,
    const int* __restrict__ srcp, const int* __restrict__ dstp,
    const float* __restrict__ W1, const float* __restrict__ b1,
    const float* __restrict__ W2, const float* __restrict__ b2,
    float* __restrict__ accum)
{
    // weights swizzled to B-fragment-friendly order: [kchunk][n][kk]
    __shared__ __align__(16) __bf16 w1s[4][64][32];   // 16 KB
    __shared__ __align__(16) __bf16 w2s[2][64][32];   // 8 KB
    // per-wave 16x64 bf16 staging tile for GEMM1 -> GEMM2 relayout (16 KB)
    __shared__ __align__(16) __bf16 stage[8][16][64];

    const int tid  = threadIdx.x;
    const int lane = tid & 31;
    const int wave = tid >> 5;
    const int nlo  = lane & 15;
    const int hi   = (lane & 16) ? 1 : 0;
    const int khi  = hi ? 16 : 0;
    const int rbase = hi ? 8 : 0;

    // cooperative fill of W1/W2 (row-major f32 -> swizzled bf16 LDS)
    for (int i = tid; i < 4 * 64 * 32; i += 256) {
        int kc  = i >> 11;
        int rem = i & 2047;
        int n   = rem >> 5;
        int kk  = rem & 31;
        w1s[kc][n][kk] = (__bf16)W1[(kc * 32 + kk) * 64 + n];
    }
    for (int i = tid; i < 2 * 64 * 32; i += 256) {
        int kc  = i >> 11;
        int rem = i & 2047;
        int n   = rem >> 5;
        int kk  = rem & 31;
        w2s[kc][n][kk] = (__bf16)W2[(kc * 32 + kk) * 64 + n];
    }
    __syncthreads();

    // W1 B-fragments: loop-invariant, in registers (16 frags x 8 VGPRs = 128)
    v16bf w1f[4][4];
#pragma unroll
    for (int kc = 0; kc < 4; ++kc)
#pragma unroll
        for (int nt = 0; nt < 4; ++nt) {
            const __bf16* wp = &w1s[kc][nt * 16 + nlo][khi];
            ((v8bf*)&w1f[kc][nt])[0] = *(const v8bf*)wp;
            ((v8bf*)&w1f[kc][nt])[1] = *(const v8bf*)(wp + 8);
        }

    float b1v[4], b2v[4];
    for (int nt = 0; nt < 4; ++nt) {
        b1v[nt] = b1[nt * 16 + nlo];
        b2v[nt] = b2[nt * 16 + nlo];
    }

    __bf16 (*st)[64] = stage[wave];
    const int m = nlo;
    const int tstride = gridDim.x * 8;
    const unsigned aoff = (unsigned)(hi ? 16 : 0);   // element offset in gathers

    int tile = blockIdx.x * 8 + wave;                // wave-uniform
    if (tile >= NTILES) return;

    // ---- prologue: indices + feature gather for the first tile ----
    int s, d;
    {
        unsigned e = (unsigned)(tile * 16 + m);
        if (e < N_EDGES) { s = srcp[e]; d = dstp[e]; }
        else             { s = (int)(e - N_EDGES); d = s; }
    }
    v16bf a[4];
#pragma unroll
    for (int c = 0; c < 4; ++c) {
        const unsigned node = (unsigned)((c < 2) ? d : s);
        const unsigned off  = (node << 6) + (unsigned)((c & 1) * 32) + aoff;
        ((v8bf*)&a[c])[0] = *(const v8bf*)(xin + off);        // K run 16h+0..7
        ((v8bf*)&a[c])[1] = *(const v8bf*)(xin + off + 8);    // K run 16h+8..15
    }

    for (; tile < NTILES; tile += tstride) {
        const int next = tile + tstride;

        // dst node ids for the 8 rows this lane covers in C-layout
        unsigned nd[8];
#pragma unroll
        for (int j = 0; j < 8; ++j)
            nd[j] = (unsigned)__shfl(d, j + rbase, 32) << 6;

        // ---- GEMM1: [16x128] @ [128x64] + b1, relu, bf16 restage ----
#pragma unroll
        for (int nt = 0; nt < 4; ++nt) {
            float bb = b1v[nt];
            v8f acc = { bb, bb, bb, bb, bb, bb, bb, bb };
#pragma unroll
            for (int kc = 0; kc < 4; ++kc)
                acc = __builtin_amdgcn_wmma_f32_16x16x32_bf16(
                    false, a[kc], false, w1f[kc][nt], (short)0, acc, false, false);
#pragma unroll
            for (int j = 0; j < 8; ++j) {
                float v = acc[j];
                v = v > 0.0f ? v : 0.0f;
                st[rbase + j][nt * 16 + nlo] = (__bf16)v;
            }
        }

        // ---- prefetch next tile: indices + feature gathers into a[] ----
        int s2 = s, d2 = d;
        if (next < NTILES) {                       // wave-uniform
            unsigned e2 = (unsigned)(next * 16 + m);
            if (e2 < N_EDGES) { s2 = srcp[e2]; d2 = dstp[e2]; }
            else              { s2 = (int)(e2 - N_EDGES); d2 = s2; }
#pragma unroll
            for (int c = 0; c < 4; ++c) {
                const unsigned node = (unsigned)((c < 2) ? d2 : s2);
                const unsigned off  = (node << 6) + (unsigned)((c & 1) * 32) + aoff;
                ((v8bf*)&a[c])[0] = *(const v8bf*)(xin + off);
                ((v8bf*)&a[c])[1] = *(const v8bf*)(xin + off + 8);
            }
        }

        // reload staging as A fragments for GEMM2 (K=64 -> 2 chunks)
        v16bf a2[2];
#pragma unroll
        for (int kc = 0; kc < 2; ++kc) {
            const __bf16* sp = &st[m][kc * 32 + (hi ? 16 : 0)];
            ((v8bf*)&a2[kc])[0] = *(const v8bf*)sp;
            ((v8bf*)&a2[kc])[1] = *(const v8bf*)(sp + 8);
        }

        // ---- GEMM2: [16x64] @ [64x64] + b2, atomic scatter into accum ----
        // W2 fragments double-buffered across nt so ds loads overlap WMMAs.
        v16bf w2cur[2], w2nxt[2];
#pragma unroll
        for (int kc = 0; kc < 2; ++kc) {
            const __bf16* wp = &w2s[kc][nlo][khi];           // nt = 0
            ((v8bf*)&w2cur[kc])[0] = *(const v8bf*)wp;
            ((v8bf*)&w2cur[kc])[1] = *(const v8bf*)(wp + 8);
        }
#pragma unroll
        for (int nt = 0; nt < 4; ++nt) {
            if (nt < 3) {
#pragma unroll
                for (int kc = 0; kc < 2; ++kc) {
                    const __bf16* wp = &w2s[kc][(nt + 1) * 16 + nlo][khi];
                    ((v8bf*)&w2nxt[kc])[0] = *(const v8bf*)wp;
                    ((v8bf*)&w2nxt[kc])[1] = *(const v8bf*)(wp + 8);
                }
            }
            float bb = b2v[nt];
            v8f a2c = { bb, bb, bb, bb, bb, bb, bb, bb };
            a2c = __builtin_amdgcn_wmma_f32_16x16x32_bf16(
                false, a2[0], false, w2cur[0], (short)0, a2c, false, false);
            a2c = __builtin_amdgcn_wmma_f32_16x16x32_bf16(
                false, a2[1], false, w2cur[1], (short)0, a2c, false, false);
            const unsigned cbase = (unsigned)(nt * 16 + nlo);
#pragma unroll
            for (int j = 0; j < 8; ++j)
                atomicAdd(accum + (nd[j] + cbase), a2c[j]);
            w2cur[0] = w2nxt[0];
            w2cur[1] = w2nxt[1];
        }

        s = s2; d = d2;
    }
}

// ---------------- pooling (LDS-staged) + final linear ----------------

#define POOL_BLOCKS 64

__global__ void __launch_bounds__(256) k_pool(const float* __restrict__ accum2,
                                              const int* __restrict__ batch,
                                              float* __restrict__ pooled,
                                              float* __restrict__ counts) {
    __shared__ float lp[N_GRAPHS * 64];
    __shared__ float lc[N_GRAPHS];

    const int t = threadIdx.x;
    for (int i = t; i < N_GRAPHS * 64; i += 256) lp[i] = 0.0f;
    if (t < N_GRAPHS) lc[t] = 0.0f;
    __syncthreads();

    const int span  = (N_NODES + POOL_BLOCKS - 1) / POOL_BLOCKS;   // 782
    const int nbeg  = blockIdx.x * span;
    int nend = nbeg + span;
    if (nend > N_NODES) nend = N_NODES;

    const int c = t & 63;
    for (int n = nbeg + (t >> 6); n < nend; n += 4) {
        float h = accum2[(unsigned)n * 64u + (unsigned)c];
        h = h > 0.0f ? h : 0.0f;
        int g = batch[n];
        atomicAdd(&lp[g * 64 + c], h);
        if (c == 0) atomicAdd(&lc[g], 1.0f);
    }
    __syncthreads();

    for (int i = t; i < N_GRAPHS * 64; i += 256) {
        float v = lp[i];
        if (v != 0.0f) atomicAdd(&pooled[i], v);
    }
    if (t < N_GRAPHS) {
        float v = lc[t];
        if (v != 0.0f) atomicAdd(&counts[t], v);
    }
}

__global__ void __launch_bounds__(256) k_final(const float* __restrict__ pooled,
                                               const float* __restrict__ counts,
                                               const float* __restrict__ Wl,
                                               const float* __restrict__ bl,
                                               float* __restrict__ out) {
    int i = blockIdx.x * 256 + threadIdx.x;
    if (i >= N_GRAPHS * 64) return;
    int g = i >> 6, o = i & 63;
    float cnt = counts[g];
    cnt = cnt < 1.0f ? 1.0f : cnt;
    float inv = 1.0f / cnt;
    float acc = bl[o];
#pragma unroll 8
    for (int k = 0; k < 64; ++k)
        acc += (pooled[g * 64 + k] * inv) * Wl[k * 64 + o];
    out[i] = acc;
}

// ---------------- launcher ----------------

extern "C" void kernel_launch(void* const* d_in, const int* in_sizes, int n_in,
                              void* d_out, int out_size, void* d_ws, size_t ws_size,
                              hipStream_t stream) {
    const float* x   = (const float*)d_in[0];
    const int*   ei  = (const int*)d_in[1];      // [2][E]: row0=src, row1=dst
    const int*   bat = (const int*)d_in[2];
    const float* W1a = (const float*)d_in[3];
    const float* b1a = (const float*)d_in[4];
    const float* W2a = (const float*)d_in[5];
    const float* b2a = (const float*)d_in[6];
    const float* W1b = (const float*)d_in[7];
    const float* b1b = (const float*)d_in[8];
    const float* W2b = (const float*)d_in[9];
    const float* b2b = (const float*)d_in[10];
    const float* Wl  = (const float*)d_in[11];
    const float* bl  = (const float*)d_in[12];

    const int* srcp = ei;
    const int* dstp = ei + N_EDGES;

    // workspace layout (all 16B-aligned by construction)
    float*  accum1 = (float*)d_ws;                       // NC f32
    float*  accum2 = accum1 + NC;                        // NC f32
    __bf16* xbf    = (__bf16*)(accum2 + NC);             // NC bf16
    __bf16* h1bf   = xbf + NC;                           // NC bf16
    float*  pooled = (float*)(h1bf + NC);                // 64*64 f32
    float*  counts = pooled + N_GRAPHS * 64;             // 64 f32

    const int gN = (NC + 255) / 256;
    const int gZ = (2 * NC + 255) / 256;

    k_zero<<<gZ, 256, 0, stream>>>(accum1, 2 * NC);
    k_zero<<<(N_GRAPHS * 65 + 255) / 256, 256, 0, stream>>>(pooled, N_GRAPHS * 65);
    k_cvt <<<gN, 256, 0, stream>>>(x, xbf, NC);

    k_edges<<<EDGE_BLOCKS, 256, 0, stream>>>(xbf, srcp, dstp, W1a, b1a, W2a, b2a, accum1);
    k_relu_cvt<<<gN, 256, 0, stream>>>(accum1, h1bf, NC);
    k_edges<<<EDGE_BLOCKS, 256, 0, stream>>>(h1bf, srcp, dstp, W1b, b1b, W2b, b2b, accum2);

    k_pool<<<POOL_BLOCKS, 256, 0, stream>>>(accum2, bat, pooled, counts);
    k_final<<<(N_GRAPHS * 64 + 255) / 256, 256, 0, stream>>>(pooled, counts, Wl, bl,
                                                             (float*)d_out);
}
